// Block_63307817943752
// MI455X (gfx1250) — 16-bit and 8-bit matrix operand order repaired and checked by reading, compiled, not run on hardware
//
#include <hip/hip_runtime.h>
#include <stdint.h>

typedef __attribute__((ext_vector_type(8)))  int      v8i;
typedef __attribute__((ext_vector_type(8)))  float    v8f;
typedef __attribute__((ext_vector_type(16))) _Float16 v16h;

union F16Frag { v16h v; unsigned u[8]; _Float16 h[16]; };
union I8Frag  { v8i  v; int i[8]; };

#define DEV static __device__ __forceinline__

DEV void async_copy_b128(void* lds, const void* gaddr) {
  unsigned l = (unsigned)(size_t)lds;
  asm volatile("global_load_async_to_lds_b128 %0, %1, off" ::"v"(l), "v"(gaddr)
               : "memory");
}
DEV void wait_async0() { asm volatile("s_wait_asynccnt 0x0" ::: "memory"); }
DEV void wait_async4() { asm volatile("s_wait_asynccnt 0x4" ::: "memory"); }

DEV float wave_sum(float v) {
  #pragma unroll
  for (int o = 16; o > 0; o >>= 1) v += __shfl_xor(v, o, 32);
  return v;
}
DEV float wave_max(float v) {
  #pragma unroll
  for (int o = 16; o > 0; o >>= 1) v = fmaxf(v, __shfl_xor(v, o, 32));
  return v;
}
DEV float half_sum(float v) {
  #pragma unroll
  for (int o = 1; o < 16; o <<= 1) v += __shfl_xor(v, o, 32);
  return v;
}
DEV float half_max(float v) {
  #pragma unroll
  for (int o = 1; o < 16; o <<= 1) v = fmaxf(v, __shfl_xor(v, o, 32));
  return v;
}
DEV float block_sum256(float v, float* sh) {
  int lane = threadIdx.x & 31, w = threadIdx.x >> 5;
  v = wave_sum(v);
  __syncthreads();
  if (lane == 0) sh[w] = v;
  __syncthreads();
  if (w == 0) {
    float r = (lane < 8) ? sh[lane] : 0.0f;
    r = wave_sum(r);
    if (lane == 0) sh[0] = r;
  }
  __syncthreads();
  return sh[0];
}
DEV float block_max256(float v, float* sh) {
  int lane = threadIdx.x & 31, w = threadIdx.x >> 5;
  v = wave_max(v);
  __syncthreads();
  if (lane == 0) sh[w] = v;
  __syncthreads();
  if (w == 0) {
    float r = (lane < 8) ? sh[lane] : -3.0e38f;
    r = wave_max(r);
    if (lane == 0) sh[0] = r;
  }
  __syncthreads();
  return sh[0];
}


__global__ __launch_bounds__(256) void abssum_kernel(const float* __restrict__ w, int n,
                                                     float* __restrict__ accum) {
  __shared__ float sh[8];
  float s = 0.0f;
  for (int i = blockIdx.x * 256 + threadIdx.x; i < n; i += gridDim.x * 256)
    s += fabsf(w[i]);
  s = block_sum256(s, sh);
  if (threadIdx.x == 0) atomicAdd(accum, s);
}

__global__ __launch_bounds__(256) void wquant_kernel(const float* __restrict__ w,
                                                     int8_t* __restrict__ wq, int n,
                                                     const float* __restrict__ accum) {
  float mean = fmaxf(accum[0] / (float)n, 1e-5f);
  float ws = 1.0f / mean;
  for (int i = blockIdx.x * 256 + threadIdx.x; i < n; i += gridDim.x * 256) {
    float q = rintf(w[i] * ws);
    q = fminf(fmaxf(q, -1.0f), 1.0f);
    wq[i] = (int8_t)(int)q;
  }
}


__global__ __launch_bounds__(256) void ln_quant_kernel(const float* __restrict__ x,
                                                       const float* __restrict__ g,
                                                       const float* __restrict__ b,
                                                       int8_t* __restrict__ outq,
                                                       float* __restrict__ arow, int D) {
  __shared__ float sh[8];
  const float* xr = x + (size_t)blockIdx.x * D;
  const int per = D / 256;
  float v[8];
  float s = 0.0f;
  for (int i = 0; i < per; i++) { v[i] = xr[threadIdx.x + i * 256]; s += v[i]; }
  float mean = block_sum256(s, sh) / (float)D;
  float s2 = 0.0f;
  for (int i = 0; i < per; i++) { float d = v[i] - mean; s2 += d * d; }
  float var = block_sum256(s2, sh) / (float)D;
  float rstd = rsqrtf(var + 1e-6f);
  float amax = 0.0f;
  for (int i = 0; i < per; i++) {
    int c = threadIdx.x + i * 256;
    v[i] = (v[i] - mean) * rstd * g[c] + b[c];
    amax = fmaxf(amax, fabsf(v[i]));
  }
  amax = fmaxf(block_max256(amax, sh), 1e-5f);
  float sc = 127.0f / amax;
  for (int i = 0; i < per; i++) {
    float q = fminf(fmaxf(rintf(v[i] * sc), -128.0f), 127.0f);
    outq[(size_t)blockIdx.x * D + threadIdx.x + i * 256] = (int8_t)(int)q;
  }
  if (threadIdx.x == 0) arow[blockIdx.x] = amax / 127.0f;
}


__global__ __launch_bounds__(256) void rowquant_kernel(const float* __restrict__ in,
                                                       int8_t* __restrict__ outq,
                                                       float* __restrict__ arow, int D) {
  __shared__ float sh[8];
  const float* r = in + (size_t)blockIdx.x * D;
  float amax = 0.0f;
  for (int i = threadIdx.x; i < D; i += 256) amax = fmaxf(amax, fabsf(r[i]));
  amax = fmaxf(block_max256(amax, sh), 1e-5f);
  float sc = 127.0f / amax;
  for (int i = threadIdx.x; i < D; i += 256) {
    float q = fminf(fmaxf(rintf(r[i] * sc), -128.0f), 127.0f);
    outq[(size_t)blockIdx.x * D + i] = (int8_t)(int)q;
  }
  if (threadIdx.x == 0) arow[blockIdx.x] = amax / 127.0f;
}


template <int OUT_F16, int RELU2, int HAS_RES>
__global__ __launch_bounds__(256) void gemm_i8_kernel(
    const int8_t* __restrict__ A, const int8_t* __restrict__ W,
    const float* __restrict__ arow, const float* __restrict__ waccum, float winv,
    const float* __restrict__ res, void* __restrict__ Cout, int M, int N, int K) {
  __shared__ alignas(16) int8_t As[2][128 * 64];
  __shared__ alignas(16) int8_t Bs[2][128 * 64];
  const int tid = threadIdx.x;
  const int lane = tid & 31, wave = tid >> 5;
  const int wm = wave & 3, wn = wave >> 2;
  const int half = lane >> 4, l16 = lane & 15;
  const int bm = blockIdx.y * 128, bn = blockIdx.x * 128;

  v8i acc[2][4];
  v8i zero8 = {0, 0, 0, 0, 0, 0, 0, 0};
  #pragma unroll
  for (int i = 0; i < 2; i++)
    #pragma unroll
    for (int j = 0; j < 4; j++) acc[i][j] = zero8;

  const int i0 = tid, i1 = tid + 256;
  const int r0 = i0 >> 2, c0 = (i0 & 3) * 16;
  const int r1 = i1 >> 2, c1 = (i1 & 3) * 16;

  async_copy_b128(&As[0][i0 * 16], A + (size_t)(bm + r0) * K + c0);
  async_copy_b128(&Bs[0][i0 * 16], W + (size_t)(bn + r0) * K + c0);
  async_copy_b128(&As[0][i1 * 16], A + (size_t)(bm + r1) * K + c1);
  async_copy_b128(&Bs[0][i1 * 16], W + (size_t)(bn + r1) * K + c1);

  const int nt = K >> 6;
  for (int j = 0; j < nt; j++) {
    const int cur = j & 1;
    if (j + 1 < nt) {
      const int nk = (j + 1) << 6;
      const int nb = cur ^ 1;
      async_copy_b128(&As[nb][i0 * 16], A + (size_t)(bm + r0) * K + nk + c0);
      async_copy_b128(&Bs[nb][i0 * 16], W + (size_t)(bn + r0) * K + nk + c0);
      async_copy_b128(&As[nb][i1 * 16], A + (size_t)(bm + r1) * K + nk + c1);
      async_copy_b128(&Bs[nb][i1 * 16], W + (size_t)(bn + r1) * K + nk + c1);
      wait_async4();
    } else {
      wait_async0();
    }
    __syncthreads();

    I8Frag af[2], bf[4];
    #pragma unroll
    for (int mt = 0; mt < 2; mt++) {
      int m = wm * 32 + mt * 16 + l16;
      #pragma unroll
      for (int v = 0; v < 8; v++) {
        int kb = ((v >> 2) << 5) + (((v >> 1) & 1) << 4) + ((v & 1) << 2) + (half << 3);
        af[mt].i[v] = *(const int*)(&As[cur][m * 64 + kb]);
      }
    }
    #pragma unroll
    for (int ntt = 0; ntt < 4; ntt++) {
      int n = wn * 64 + ntt * 16 + l16;
      #pragma unroll
      for (int v = 0; v < 8; v++) {
        int kb = ((v >> 2) << 5) + (((v >> 1) & 1) << 4) + ((v & 1) << 2) + (half << 3);
        bf[ntt].i[v] = *(const int*)(&Bs[cur][n * 64 + kb]);
      }
    }
    #pragma unroll
    for (int mt = 0; mt < 2; mt++)
      #pragma unroll
      for (int ntt = 0; ntt < 4; ntt++)
        acc[mt][ntt] = __builtin_amdgcn_wmma_i32_16x16x64_iu8(
            true, af[mt].v, true, bf[ntt].v, acc[mt][ntt], false, false);
    __syncthreads();
  }

  float wmul = fmaxf(waccum[0] * winv, 1e-5f);
  #pragma unroll
  for (int mt = 0; mt < 2; mt++) {
    #pragma unroll
    for (int v = 0; v < 8; v++) {
      int row = bm + wm * 32 + mt * 16 + v + half * 8;
      float am = arow[row] * wmul;
      #pragma unroll
      for (int ntt = 0; ntt < 4; ntt++) {
        int col = bn + wn * 64 + ntt * 16 + l16;
        float val = (float)acc[mt][ntt][v] * am;
        if (RELU2) { val = fmaxf(val, 0.0f); val = val * val; }
        if (HAS_RES) val += res[(size_t)row * N + col];
        if (OUT_F16)
          ((_Float16*)Cout)[(size_t)row * N + col] = (_Float16)val;
        else
          ((float*)Cout)[(size_t)row * N + col] = val;
      }
    }
  }
}


__global__ __launch_bounds__(128) void attn_kernel(const _Float16* __restrict__ Q,
                                                   const _Float16* __restrict__ Kg,
                                                   const _Float16* __restrict__ Vg,
                                                   float* __restrict__ O, int T) {
  const int head = blockIdx.y;
  const int kvh = head >> 2;
  const int qbase = blockIdx.x * 64;
  const int lane = threadIdx.x & 31, wave = threadIdx.x >> 5;
  const int half = lane >> 4, l16 = lane & 15;
  const int qrow0 = qbase + wave * 16;

  __shared__ alignas(16) _Float16 Ks[32 * 64];
  __shared__ alignas(16) _Float16 Vs[32 * 64];
  __shared__ alignas(4) _Float16 Ps[4][16 * 32];

  F16Frag qf[2];
  {
    const _Float16* qp = Q + (size_t)(qrow0 + l16) * 2048 + head * 64;
    #pragma unroll
    for (int f = 0; f < 2; f++)
      #pragma unroll
      for (int v = 0; v < 8; v++) {
        int kp = ((v >> 2) << 4) + ((v & 3) << 1) + (half << 3);
        qf[f].u[v] = *(const unsigned*)(qp + f * 32 + kp);
      }
  }

  v8f ao[4];
  float mrow[8], lrow[8];
  #pragma unroll
  for (int t = 0; t < 4; t++)
    #pragma unroll
    for (int e = 0; e < 8; e++) ao[t][e] = 0.0f;
  #pragma unroll
  for (int e = 0; e < 8; e++) { mrow[e] = -3.0e38f; lrow[e] = 0.0f; }

  const int ti0 = threadIdx.x, ti1 = threadIdx.x + 128;
  const int tr0 = ti0 >> 3, tc0 = (ti0 & 7) * 8;
  const int tr1 = ti1 >> 3, tc1 = (ti1 & 7) * 8;

  const int nkt = qbase / 32 + 2;
  for (int j = 0; j < nkt; j++) {
    const int kb = j * 32;
    async_copy_b128(&Ks[ti0 * 8], Kg + (size_t)(kb + tr0) * 512 + kvh * 64 + tc0);
    async_copy_b128(&Vs[ti0 * 8], Vg + (size_t)(kb + tr0) * 512 + kvh * 64 + tc0);
    async_copy_b128(&Ks[ti1 * 8], Kg + (size_t)(kb + tr1) * 512 + kvh * 64 + tc1);
    async_copy_b128(&Vs[ti1 * 8], Vg + (size_t)(kb + tr1) * 512 + kvh * 64 + tc1);
    wait_async0();
    __syncthreads();

    if (kb <= qrow0 + 15) {
      v8f st[2];
      #pragma unroll
      for (int nt = 0; nt < 2; nt++) {
        F16Frag b0, b1;
        int n = nt * 16 + l16;
        #pragma unroll
        for (int v = 0; v < 8; v++) {
          int kp = ((v >> 2) << 4) + ((v & 3) << 1) + (half << 3);
          b0.u[v] = *(const unsigned*)(Ks + n * 64 + kp);
          b1.u[v] = *(const unsigned*)(Ks + n * 64 + 32 + kp);
        }
        v8f z;
        #pragma unroll
        for (int e = 0; e < 8; e++) z[e] = 0.0f;
        v8f s = __builtin_amdgcn_wmma_f32_16x16x32_f16(false, qf[0].v, false, b0.v,
                                                       (short)0, z, false, false);
        s = __builtin_amdgcn_wmma_f32_16x16x32_f16(false, qf[1].v, false, b1.v,
                                                   (short)0, s, false, false);
        st[nt] = s;
      }
      #pragma unroll
      for (int v = 0; v < 8; v++) {
        int qi = qrow0 + v + half * 8;
        float s0 = st[0][v] * 0.125f;
        float s1 = st[1][v] * 0.125f;
        if (kb + l16 > qi) s0 = -3.0e38f;
        if (kb + 16 + l16 > qi) s1 = -3.0e38f;
        float rm = half_max(fmaxf(s0, s1));
        float mn = fmaxf(mrow[v], rm);
        float alpha = __expf(mrow[v] - mn);
        float p0 = __expf(s0 - mn);
        float p1 = __expf(s1 - mn);
        float rs = half_sum(p0 + p1);
        lrow[v] = lrow[v] * alpha + rs;
        mrow[v] = mn;
        #pragma unroll
        for (int t = 0; t < 4; t++) ao[t][v] *= alpha;
        int m = v + half * 8;
        Ps[wave][m * 32 + l16] = (_Float16)p0;
        Ps[wave][m * 32 + 16 + l16] = (_Float16)p1;
      }
      F16Frag pf;
      {
        const _Float16* pp = Ps[wave];
        #pragma unroll
        for (int v = 0; v < 8; v++) {
          int kp = ((v >> 2) << 4) + ((v & 3) << 1) + (half << 3);
          pf.u[v] = *(const unsigned*)(pp + l16 * 32 + kp);
        }
      }
      #pragma unroll
      for (int ot = 0; ot < 4; ot++) {
        F16Frag vb;
        int d = ot * 16 + l16;
        #pragma unroll
        for (int v = 0; v < 8; v++) {
          int kk = ((v >> 2) << 4) + ((v & 3) << 1) + (half << 3);
          vb.h[2 * v] = Vs[kk * 64 + d];
          vb.h[2 * v + 1] = Vs[(kk + 1) * 64 + d];
        }
        ao[ot] = __builtin_amdgcn_wmma_f32_16x16x32_f16(false, pf.v, false, vb.v,
                                                        (short)0, ao[ot], false, false);
      }
    }
    __syncthreads();
  }

  #pragma unroll
  for (int v = 0; v < 8; v++) {
    int row = qrow0 + v + half * 8;
    float linv = 1.0f / lrow[v];
    #pragma unroll
    for (int ot = 0; ot < 4; ot++) {
      int d = ot * 16 + l16;
      O[(size_t)row * 2048 + head * 64 + d] = ao[ot][v] * linv;
    }
  }
}


static inline size_t align256(size_t x) { return (x + 255) & ~(size_t)255; }

extern "C" void kernel_launch(void* const* d_in, const int* in_sizes, int n_in,
                              void* d_out, int out_size, void* d_ws, size_t ws_size,
                              hipStream_t stream) {
  (void)in_sizes; (void)n_in; (void)out_size; (void)ws_size;
  const float* x  = (const float*)d_in[0];
  const float* g1 = (const float*)d_in[1];
  const float* b1 = (const float*)d_in[2];
  const float* g2 = (const float*)d_in[3];
  const float* b2 = (const float*)d_in[4];
  const float* wq = (const float*)d_in[5];
  const float* wk = (const float*)d_in[6];
  const float* wv = (const float*)d_in[7];
  const float* wo = (const float*)d_in[8];
  const float* w1 = (const float*)d_in[9];
  const float* w2 = (const float*)d_in[10];
  float* out = (float*)d_out;

  const int T = 2048, D = 2048, KV = 512, F = 8192;

  char* ws = (char*)d_ws;
  size_t off = 0;
  auto alloc = [&](size_t bytes) { char* p = ws + off; off = align256(off + bytes); return p; };
  float*    accum = (float*)alloc(8 * sizeof(float));
  int8_t*   wq8 = (int8_t*)alloc((size_t)D * D);
  int8_t*   wk8 = (int8_t*)alloc((size_t)KV * D);
  int8_t*   wv8 = (int8_t*)alloc((size_t)KV * D);
  int8_t*   wo8 = (int8_t*)alloc((size_t)D * D);
  int8_t*   w18 = (int8_t*)alloc((size_t)F * D);
  int8_t*   w28 = (int8_t*)alloc((size_t)D * F);
  int8_t*   hq  = (int8_t*)alloc((size_t)T * D);
  float*    asc1 = (float*)alloc(T * sizeof(float));
  _Float16* qh = (_Float16*)alloc((size_t)T * D * 2);
  _Float16* kh = (_Float16*)alloc((size_t)T * KV * 2);
  _Float16* vh = (_Float16*)alloc((size_t)T * KV * 2);
  float*    o32 = (float*)alloc((size_t)T * D * 4);
  int8_t*   oq  = (int8_t*)alloc((size_t)T * D);
  float*    osc = (float*)alloc(T * sizeof(float));
  float*    x1  = (float*)alloc((size_t)T * D * 4);
  int8_t*   h2q = (int8_t*)alloc((size_t)T * D);
  float*    h2sc = (float*)alloc(T * sizeof(float));
  float*    fbuf = (float*)alloc((size_t)T * F * 4);
  int8_t*   fq   = (int8_t*)alloc((size_t)T * F);
  float*    fsc  = (float*)alloc(T * sizeof(float));

  hipMemsetAsync(accum, 0, 8 * sizeof(float), stream);

  const float* wsrc[6] = {wq, wk, wv, wo, w1, w2};
  int8_t* wdst[6] = {wq8, wk8, wv8, wo8, w18, w28};
  const int wcnt[6] = {D * D, KV * D, KV * D, D * D, F * D, D * F};
  for (int i = 0; i < 6; i++)
    abssum_kernel<<<1024, 256, 0, stream>>>(wsrc[i], wcnt[i], accum + i);
  for (int i = 0; i < 6; i++)
    wquant_kernel<<<1024, 256, 0, stream>>>(wsrc[i], wdst[i], wcnt[i], accum + i);

  ln_quant_kernel<<<T, 256, 0, stream>>>(x, g1, b1, hq, asc1, D);

  dim3 blk(256);
  gemm_i8_kernel<1, 0, 0><<<dim3(D / 128, T / 128), blk, 0, stream>>>(
      hq, wq8, asc1, accum + 0, 1.0f / (float)(D * D), nullptr, qh, T, D, D);
  gemm_i8_kernel<1, 0, 0><<<dim3(KV / 128, T / 128), blk, 0, stream>>>(
      hq, wk8, asc1, accum + 1, 1.0f / (float)(KV * D), nullptr, kh, T, KV, D);
  gemm_i8_kernel<1, 0, 0><<<dim3(KV / 128, T / 128), blk, 0, stream>>>(
      hq, wv8, asc1, accum + 2, 1.0f / (float)(KV * D), nullptr, vh, T, KV, D);

  attn_kernel<<<dim3(T / 64, 32), dim3(128), 0, stream>>>(qh, kh, vh, o32, T);

  rowquant_kernel<<<T, 256, 0, stream>>>(o32, oq, osc, D);
  gemm_i8_kernel<0, 0, 1><<<dim3(D / 128, T / 128), blk, 0, stream>>>(
      oq, wo8, osc, accum + 3, 1.0f / (float)(D * D), x, x1, T, D, D);

  ln_quant_kernel<<<T, 256, 0, stream>>>(x1, g2, b2, h2q, h2sc, D);
  gemm_i8_kernel<0, 1, 0><<<dim3(F / 128, T / 128), blk, 0, stream>>>(
      h2q, w18, h2sc, accum + 4, 1.0f / (float)(F * D), nullptr, fbuf, T, F, D);
  rowquant_kernel<<<T, 256, 0, stream>>>(fbuf, fq, fsc, F);
  gemm_i8_kernel<0, 0, 1><<<dim3(D / 128, T / 128), blk, 0, stream>>>(
      fq, w28, fsc, accum + 5, 1.0f / (float)(D * F), x1, out, T, D, F);
}
